// HAT_47596827574583
// MI455X (gfx1250) — matrix operand order repaired and checked by reading, not compiled, not run on hardware
//
#include <hip/hip_runtime.h>
#include <hip/hip_bf16.h>
#include <math.h>

#define D 128

typedef __attribute__((ext_vector_type(16))) _Float16 v16h;
typedef __attribute__((ext_vector_type(8)))  _Float16 v8h;
typedef __attribute__((ext_vector_type(4)))  _Float16 v4h;
typedef __attribute__((ext_vector_type(8)))  float    v8f;

__device__ __forceinline__ void atomic_max_float(float* addr, float val) {
    // classic bit-pattern trick: int-max for >=0, uint-min for <0; init must be -inf
    if (val >= 0.0f) atomicMax((int*)addr, __float_as_int(val));
    else             atomicMin((unsigned int*)addr, __float_as_uint(val));
}

// ---------- weights f32 -> f16, packed [L][3][D][D] ----------
__global__ __launch_bounds__(256) void wconv_kernel(
    const float* __restrict__ Wq, const float* __restrict__ Wk,
    const float* __restrict__ Wv, _Float16* __restrict__ W16, int total) {
    int i = blockIdx.x * blockDim.x + threadIdx.x;
    if (i >= total) return;
    const int dd = D * D;
    int l   = i / (3 * dd);
    int mat = (i / dd) % 3;
    int r   = i % dd;
    const float* Wsrc = (mat == 0) ? Wq : (mat == 1) ? Wk : Wv;
    W16[i] = (_Float16)Wsrc[(size_t)l * dd + r];
}

// ---------- log_map at origin: t = (2/sc)*atanh(sc*|x|)*x/|x|, f16 out ----------
__global__ __launch_bounds__(256) void logmap_kernel(
    const float* __restrict__ x, _Float16* __restrict__ t16,
    const float* __restrict__ curv, int N) {
    int gid  = blockIdx.x * blockDim.x + threadIdx.x;
    int node = gid >> 5;
    int lane = gid & 31;
    if (node >= N) return;
    float4 xv = ((const float4*)(x + (size_t)node * D))[lane];
    float ss = xv.x * xv.x + xv.y * xv.y + xv.z * xv.z + xv.w * xv.w;
#pragma unroll
    for (int off = 16; off > 0; off >>= 1) ss += __shfl_xor(ss, off, 32);
    float sc = sqrtf(curv[0]);
    float n  = sqrtf(ss);
    float f  = (n > 0.0f) ? (2.0f / sc) * atanhf(sc * n) / n : 2.0f;
    v4h o;
    o[0] = (_Float16)(f * xv.x);
    o[1] = (_Float16)(f * xv.y);
    o[2] = (_Float16)(f * xv.z);
    o[3] = (_Float16)(f * xv.w);
    *(v4h*)(t16 + (size_t)node * D + lane * 4) = o;
}

// ---------- init m=-inf, z=0, h=0 ----------
__global__ __launch_bounds__(256) void init_kernel(
    float* __restrict__ m, float* __restrict__ z, float* __restrict__ h, int N) {
    int i = blockIdx.x * blockDim.x + threadIdx.x;
    if (i < N * D) h[i] = 0.0f;
    if (i < N) { m[i] = -INFINITY; z[i] = 0.0f; }
}

// ---------- q/k/v projection via v_wmma_f32_16x16x32_f16 ----------
// grid: (N/16 tiles, D/64 feature strips, 3 matrices); 1 wave per block.
// Each wave computes a 16x64 output strip: A tile loaded once per K step,
// reused across 4 B tiles -> 16 WMMAs/wave, 4 independent accumulator chains.
__global__ __launch_bounds__(32) void qkv_gemm_kernel(
    const _Float16* __restrict__ t16, const _Float16* __restrict__ W16,
    const float* __restrict__ bq, const float* __restrict__ bk,
    const float* __restrict__ bv,
    float* __restrict__ q, float* __restrict__ k, float* __restrict__ v, int N) {
    int lane = threadIdx.x;
    int l    = lane & 15;
    int hi   = lane >> 4;                 // half-wave select
    int ntile = blockIdx.x, jgroup = blockIdx.y, mat = blockIdx.z;

    const _Float16* W    = W16 + (size_t)mat * D * D;
    const float*    bias = (mat == 0) ? bq : (mat == 1) ? bk : bv;
    float*          out  = (mat == 0) ? q : (mat == 1) ? k : v;

    int row = ntile * 16 + l; if (row >= N) row = N - 1;   // clamp (EXEC stays all-1)
    int jbase = jgroup * 64;

    // A (16x32 MxK, f16): lanes 0-15 row=l hold K 0..15; lanes 16-31 K 16..31
    const _Float16* abase = t16 + (size_t)row * D + (hi << 4);
    // B (32x16 KxN, f16): col N = l; lower half-wave K 0..15, upper K 16..31.
    // B[k][n] = W[n][k] with W row-major [out][in] -> 16 contiguous halfs per lane.
    const _Float16* bbase = W + (size_t)(jbase + l) * D + (hi << 4);

    v8f acc0 = {}, acc1 = {}, acc2 = {}, acc3 = {};
#pragma unroll
    for (int kb = 0; kb < D; kb += 32) {
        v16h a;
        ((v8h*)&a)[0] = *(const v8h*)(abase + kb);        // K chunk 0
        ((v8h*)&a)[1] = *(const v8h*)(abase + kb + 8);    // K chunk 1
        v16h b0 = *(const v16h*)(bbase + kb);
        v16h b1 = *(const v16h*)(bbase + 16 * D + kb);
        v16h b2 = *(const v16h*)(bbase + 32 * D + kb);
        v16h b3 = *(const v16h*)(bbase + 48 * D + kb);
        acc0 = __builtin_amdgcn_wmma_f32_16x16x32_f16(false, a, false, b0, (short)0, acc0, false, false);
        acc1 = __builtin_amdgcn_wmma_f32_16x16x32_f16(false, a, false, b1, (short)0, acc1, false, false);
        acc2 = __builtin_amdgcn_wmma_f32_16x16x32_f16(false, a, false, b2, (short)0, acc2, false, false);
        acc3 = __builtin_amdgcn_wmma_f32_16x16x32_f16(false, a, false, b3, (short)0, acc3, false, false);
    }
    float b0v = bias[jbase + l];
    float b1v = bias[jbase + 16 + l];
    float b2v = bias[jbase + 32 + l];
    float b3v = bias[jbase + 48 + l];
    int node0 = ntile * 16;
    if (node0 + 16 <= N) {
        // full tile: uniform (scalar) branch, unguarded stores
#pragma unroll
        for (int r = 0; r < 8; ++r) {                      // C/D: VGPR r -> M = r + 8*hi
            size_t ro = (size_t)(node0 + r + (hi << 3)) * D + jbase + l;
            out[ro]      = acc0[r] + b0v;
            out[ro + 16] = acc1[r] + b1v;
            out[ro + 32] = acc2[r] + b2v;
            out[ro + 48] = acc3[r] + b3v;
        }
    } else {
#pragma unroll
        for (int r = 0; r < 8; ++r) {
            int nd = node0 + r + (hi << 3);
            if (nd < N) {
                size_t ro = (size_t)nd * D + jbase + l;
                out[ro]      = acc0[r] + b0v;
                out[ro + 16] = acc1[r] + b1v;
                out[ro + 32] = acc2[r] + b2v;
                out[ro + 48] = acc3[r] + b3v;
            }
        }
    }
}

// ---------- per-edge score + atomic segment max (wave per edge) ----------
__global__ __launch_bounds__(256) void edge_score_kernel(
    const float* __restrict__ q, const float* __restrict__ k,
    const int* __restrict__ src, const int* __restrict__ dst,
    float* __restrict__ e, float* __restrict__ m, int E, float inv_scale) {
    int gid  = blockIdx.x * blockDim.x + threadIdx.x;
    int edge = gid >> 5, lane = gid & 31;
    if (edge >= E) return;
    int s = src[edge], d = dst[edge];
    float4 kv = ((const float4*)(k + (size_t)s * D))[lane];
    float4 qv = ((const float4*)(q + (size_t)d * D))[lane];
    float dot = kv.x * qv.x + kv.y * qv.y + kv.z * qv.z + kv.w * qv.w;
#pragma unroll
    for (int off = 16; off > 0; off >>= 1) dot += __shfl_xor(dot, off, 32);
    if (lane == 0) {
        float ev = dot * inv_scale;
        e[edge] = ev;
        atomic_max_float(m + d, ev);
    }
}

// ---------- p = exp(e-m[dst]); z[dst]+=p; h[dst]+= p*v[src] (unnormalized) ----------
__global__ __launch_bounds__(256) void edge_accum_kernel(
    const float* __restrict__ e, const float* __restrict__ m,
    const float* __restrict__ v,
    const int* __restrict__ src, const int* __restrict__ dst,
    float* __restrict__ z, float* __restrict__ h, int E) {
    int gid  = blockIdx.x * blockDim.x + threadIdx.x;
    int edge = gid >> 5, lane = gid & 31;
    if (edge >= E) return;
    int s = src[edge], d = dst[edge];
    float p = __expf(e[edge] - m[d]);
    if (lane == 0) atomicAdd(z + d, p);
    float4 vv = ((const float4*)(v + (size_t)s * D))[lane];
    float* hp = h + (size_t)d * D + lane * 4;
    atomicAdd(hp + 0, p * vv.x);
    atomicAdd(hp + 1, p * vv.y);
    atomicAdd(hp + 2, p * vv.z);
    atomicAdd(hp + 3, p * vv.w);
}

// ---------- exp_map at origin of h/z: x = tanh(sc*|h|/2)*h/(sc*|h|) ----------
__global__ __launch_bounds__(256) void expmap_kernel(
    const float* __restrict__ h, const float* __restrict__ z,
    float* __restrict__ xout, const float* __restrict__ curv, int N) {
    int gid  = blockIdx.x * blockDim.x + threadIdx.x;
    int node = gid >> 5, lane = gid & 31;
    if (node >= N) return;
    float zi = 1.0f / z[node];                 // fold softmax normalization in here
    float4 hv = ((const float4*)(h + (size_t)node * D))[lane];
    hv.x *= zi; hv.y *= zi; hv.z *= zi; hv.w *= zi;
    float ss = hv.x * hv.x + hv.y * hv.y + hv.z * hv.z + hv.w * hv.w;
#pragma unroll
    for (int off = 16; off > 0; off >>= 1) ss += __shfl_xor(ss, off, 32);
    float sc = sqrtf(curv[0]);
    float n  = sqrtf(ss);
    float f  = (n > 0.0f) ? tanhf(sc * n * 0.5f) / (sc * n) : 0.5f;
    float4 o; o.x = f * hv.x; o.y = f * hv.y; o.z = f * hv.z; o.w = f * hv.w;
    ((float4*)(xout + (size_t)node * D))[lane] = o;
}

extern "C" void kernel_launch(void* const* d_in, const int* in_sizes, int n_in,
                              void* d_out, int out_size, void* d_ws, size_t ws_size,
                              hipStream_t stream) {
    const float* emb  = (const float*)d_in[0];
    const float* Wq   = (const float*)d_in[1];
    const float* bq   = (const float*)d_in[2];
    const float* Wk   = (const float*)d_in[3];
    const float* bk   = (const float*)d_in[4];
    const float* Wv   = (const float*)d_in[5];
    const float* bv   = (const float*)d_in[6];
    const float* curv = (const float*)d_in[7];
    const int*   src  = (const int*)d_in[8];
    const int*   dst  = (const int*)d_in[9];

    const int N = in_sizes[0] / D;          // nodes
    const int L = in_sizes[2] / D;          // layers (bq is [L,D])
    const int E = in_sizes[8];              // edges

    // workspace carve-out (256B aligned)
    char* base = (char*)d_ws;
    auto carve = [&](size_t bytes) -> char* {
        char* p = base;
        base += (bytes + 255) & ~(size_t)255;
        return p;
    };
    _Float16* t16 = (_Float16*)carve((size_t)N * D * sizeof(_Float16));
    float*    q   = (float*)carve((size_t)N * D * sizeof(float));
    float*    k   = (float*)carve((size_t)N * D * sizeof(float));
    float*    v   = (float*)carve((size_t)N * D * sizeof(float));
    float*    h   = (float*)carve((size_t)N * D * sizeof(float));
    float*    m   = (float*)carve((size_t)N * sizeof(float));
    float*    z   = (float*)carve((size_t)N * sizeof(float));
    float*    e   = (float*)carve((size_t)E * sizeof(float));
    _Float16* W16 = (_Float16*)carve((size_t)L * 3 * D * D * sizeof(_Float16));

    float* xout = (float*)d_out;

    int wtot = L * 3 * D * D;
    wconv_kernel<<<(wtot + 255) / 256, 256, 0, stream>>>(Wq, Wk, Wv, W16, wtot);

    const float inv_scale = 1.0f / sqrtf((float)D);
    int nwave_blocks = (int)(((size_t)N * 32 + 255) / 256);
    int ewave_blocks = (int)(((size_t)E * 32 + 255) / 256);
    int init_blocks  = (int)(((size_t)N * D + 255) / 256);

    const float* xin = emb;
    for (int l = 0; l < L; ++l) {
        logmap_kernel<<<nwave_blocks, 256, 0, stream>>>(xin, t16, curv, N);
        init_kernel<<<init_blocks, 256, 0, stream>>>(m, z, h, N);
        qkv_gemm_kernel<<<dim3((N + 15) / 16, D / 64, 3), 32, 0, stream>>>(
            t16, W16 + (size_t)l * 3 * D * D,
            bq + (size_t)l * D, bk + (size_t)l * D, bv + (size_t)l * D,
            q, k, v, N);
        edge_score_kernel<<<ewave_blocks, 256, 0, stream>>>(q, k, src, dst, e, m, E, inv_scale);
        edge_accum_kernel<<<ewave_blocks, 256, 0, stream>>>(e, m, v, src, dst, z, h, E);
        expmap_kernel<<<nwave_blocks, 256, 0, stream>>>(h, z, xout, curv, N);
        xin = xout;   // d_out doubles as the inter-layer x buffer
    }
}
